// scaled_dot_product_attention_11141145166223
// MI455X (gfx1250) — compile-verified, matrix operand order checked by reading, not run on hardware
//
#include <hip/hip_runtime.h>
#include <hip/hip_bf16.h>

typedef __attribute__((ext_vector_type(16))) _Float16 v16h;
typedef __attribute__((ext_vector_type(8)))  _Float16 v8h;
typedef __attribute__((ext_vector_type(8)))  float    v8f;
typedef __attribute__((ext_vector_type(4)))  float    v4f;

#define B_ 8
#define N_ 4096
#define D_ 64
#define NEG_INF_ (-1.0e9f)

// ---------------------------------------------------------------------------
// Prep: fp32 -> f16 copies of Q,K (row major) and V transposed to [B, D, N].
// One grid-stride elementwise pass; ~25MB read + 12MB write, ~1.5us at HBM BW.
// ---------------------------------------------------------------------------
__global__ void prep_kernel(const float* __restrict__ Q, const float* __restrict__ K,
                            const float* __restrict__ V, _Float16* __restrict__ Qh,
                            _Float16* __restrict__ Kh, _Float16* __restrict__ VTh) {
    int i = blockIdx.x * blockDim.x + threadIdx.x;
    const int total = B_ * N_ * D_;
    if (i < total) {
        Qh[i] = (_Float16)Q[i];
        Kh[i] = (_Float16)K[i];
        int d = i & (D_ - 1);
        int n = (i >> 6) & (N_ - 1);
        int b = i >> 18;
        VTh[(((size_t)(b * D_ + d)) << 12) + n] = (_Float16)V[i];
    }
}

// Zero the reg_atten scalar slot (atomicAdd target) each call -> deterministic
// across graph replays.
__global__ void zero_reg_kernel(float* __restrict__ out) {
    out[(size_t)B_ * N_ * D_] = 0.f;
}

// ---------------------------------------------------------------------------
// Flash attention, one wave32 per 32 queries of one batch.
//   S^T = K . Q^T    (M=keys, N=queries)  -> lane owns a query column
//   P   = exp(S^T - m) lands directly in WMMA A/B layout (no shuffles)
//   O^T = V^T . P^T  (M=d,    N=queries)  -> per-query rescale is per-lane
// Hot-loop cross-lane traffic: ONE ds-permute per subtile-chunk (the max);
// the row-sum is accumulated per lane-half (corr is identical on both
// halves) and combined once at the end.
// ---------------------------------------------------------------------------
__launch_bounds__(32)
__global__ void flash_attn_kernel(const _Float16* __restrict__ Qh,
                                  const _Float16* __restrict__ Kh,
                                  const _Float16* __restrict__ VTh,
                                  float* __restrict__ out) {
    const int lane = threadIdx.x;        // 0..31
    const int wid  = blockIdx.x;         // B * N/32 waves
    const int b    = wid >> 7;           // 128 waves per batch
    const int q0   = (wid & 127) << 5;   // 32 queries per wave
    const int half = lane >> 4;          // lane half (A/B layout K split)
    const int l15  = lane & 15;
    const int koff = half << 3;          // +8 element offset for upper lanes

    const _Float16* Qb = Qh  + ((size_t)b * N_) * D_;
    const _Float16* Kb = Kh  + ((size_t)b * N_) * D_;
    const _Float16* Vb = VTh + ((size_t)b * D_) * N_;

    // Q as WMMA B-tiles (lane = query): [subtile s][d-chunk c], held all loop.
    v16h qt[2][2];
    #pragma unroll
    for (int s = 0; s < 2; ++s) {
        int row = q0 + s * 16 + l15;
        #pragma unroll
        for (int c = 0; c < 2; ++c) {
            int dbase = c * 32 + koff;
            v8h lo = *(const v8h*)(Qb + row * D_ + dbase);
            v8h hi = *(const v8h*)(Qb + row * D_ + dbase + 16);
            v16h t;
            #pragma unroll
            for (int i = 0; i < 8; ++i) { t[i] = lo[i]; t[i + 8] = hi[i]; }
            qt[s][c] = t;
        }
    }

    v8f o[2][4];                          // O^T accumulators [subtile][d-tile]
    #pragma unroll
    for (int s = 0; s < 2; ++s)
        #pragma unroll
        for (int t = 0; t < 4; ++t) o[s][t] = {};

    float mrun[2] = {-1e30f, -1e30f};
    float lsum[2] = {0.f, 0.f};           // per-lane-HALF partial denominators
    float reg = 0.f;

    for (int kv = 0; kv < N_; kv += 32) {
        // K rows as WMMA A-tiles: [key-tile t][d-chunk c]
        v16h ka[2][2];
        #pragma unroll
        for (int t = 0; t < 2; ++t) {
            int row = kv + t * 16 + l15;
            #pragma unroll
            for (int c = 0; c < 2; ++c) {
                int dbase = c * 32 + koff;
                v8h lo = *(const v8h*)(Kb + row * D_ + dbase);
                v8h hi = *(const v8h*)(Kb + row * D_ + dbase + 16);
                v16h x;
                #pragma unroll
                for (int i = 0; i < 8; ++i) { x[i] = lo[i]; x[i + 8] = hi[i]; }
                ka[t][c] = x;
            }
        }
        // V^T rows as WMMA A-tiles: [d-tile t], shared by both query subtiles
        v16h va[4];
        #pragma unroll
        for (int t = 0; t < 4; ++t) {
            const _Float16* p = Vb + (t * 16 + l15) * N_ + kv + koff;
            v8h lo = *(const v8h*)(p);
            v8h hi = *(const v8h*)(p + 16);
            v16h x;
            #pragma unroll
            for (int i = 0; i < 8; ++i) { x[i] = lo[i]; x[i + 8] = hi[i]; }
            va[t] = x;
        }

        // Prefetch next chunk's K (32 rows x 128B) and V^T (64 rows x 64B)
        // into the near caches while this chunk computes. No VGPR cost.
        if (kv + 32 < N_) {
            __builtin_prefetch(Kb + (size_t)(kv + 32 + lane) * D_, 0, 3);
            __builtin_prefetch(Vb + (size_t)lane * N_ + kv + 32, 0, 3);
            __builtin_prefetch(Vb + (size_t)(lane + 32) * N_ + kv + 32, 0, 3);
        }

        #pragma unroll
        for (int s = 0; s < 2; ++s) {
            // S^T tiles for keys [kv,kv+16) and [kv+16,kv+32), K-dim = D = 2x32
            v8f st0 = {}, st1 = {};
            st0 = __builtin_amdgcn_wmma_f32_16x16x32_f16(false, ka[0][0], false, qt[s][0], (short)0, st0, false, false);
            st0 = __builtin_amdgcn_wmma_f32_16x16x32_f16(false, ka[0][1], false, qt[s][1], (short)0, st0, false, false);
            st1 = __builtin_amdgcn_wmma_f32_16x16x32_f16(false, ka[1][0], false, qt[s][0], (short)0, st1, false, false);
            st1 = __builtin_amdgcn_wmma_f32_16x16x32_f16(false, ka[1][1], false, qt[s][1], (short)0, st1, false, false);

            // reg stat BEFORE mask (matches reference)
            #pragma unroll
            for (int v = 0; v < 8; ++v) reg += st0[v] * st0[v] + st1[v] * st1[v];

            // diagonal self-mask: only the kv == q0 chunk can hit the diagonal
            // (query block and key chunk are both 32-aligned). Wave-uniform
            // branch -> cold block off the hot path.
            if (kv == q0) {
                int qi = q0 + s * 16 + l15;
                #pragma unroll
                for (int v = 0; v < 8; ++v) {
                    int j0 = kv + koff + v;
                    if (qi == j0)      st0[v] = NEG_INF_;
                    if (qi == j0 + 16) st1[v] = NEG_INF_;
                }
            }

            // online softmax: per-lane partial max + one partner-lane exchange
            // (the ONLY cross-lane op in the hot loop)
            float mc = st0[0];
            #pragma unroll
            for (int v = 1; v < 8; ++v) mc = fmaxf(mc, st0[v]);
            #pragma unroll
            for (int v = 0; v < 8; ++v) mc = fmaxf(mc, st1[v]);
            mc = fmaxf(mc, __shfl_xor(mc, 16, 32));
            float mnew = fmaxf(mrun[s], mc);
            float corr = __expf(mrun[s] - mnew);
            mrun[s] = mnew;

            // P = exp(S^T - m): lands exactly in the f16 A/B operand layout.
            // lsum stays half-row-local: corr is identical on both lane halves,
            // so the halves combine linearly after the loop.
            v16h p;
            float ps = 0.f;
            #pragma unroll
            for (int v = 0; v < 8; ++v) {
                float e0 = __expf(st0[v] - mnew);
                float e1 = __expf(st1[v] - mnew);
                ps += e0 + e1;
                p[v]     = (_Float16)e0;
                p[v + 8] = (_Float16)e1;
            }
            lsum[s] = lsum[s] * corr + ps;

            // rescale O^T (per-lane scalar!) and accumulate O^T += V^T * P^T
            #pragma unroll
            for (int t = 0; t < 4; ++t) {
                v8f ot = o[s][t];
                #pragma unroll
                for (int v = 0; v < 8; ++v) ot[v] *= corr;
                o[s][t] = __builtin_amdgcn_wmma_f32_16x16x32_f16(false, va[t], false, p, (short)0, ot, false, false);
            }
        }
    }

    // finalize: combine half-row denominators, O /= l, store context.
    // The 8 lane values of each O^T tile are 8 consecutive, 32B-aligned
    // floats of one query row -> 2x b128 stores per tile.
    #pragma unroll
    for (int s = 0; s < 2; ++s) {
        float lfull = lsum[s] + __shfl_xor(lsum[s], 16, 32);
        float inv = 1.0f / lfull;
        int q = q0 + s * 16 + l15;
        float* orow = out + ((size_t)b * N_ + q) * D_;
        #pragma unroll
        for (int t = 0; t < 4; ++t) {
            v4f lo, hi;
            #pragma unroll
            for (int v = 0; v < 4; ++v) {
                lo[v] = o[s][t][v]     * inv;
                hi[v] = o[s][t][v + 4] * inv;
            }
            *(v4f*)(orow + t * 16 + koff)     = lo;
            *(v4f*)(orow + t * 16 + koff + 4) = hi;
        }
    }

    // reg_atten: wave reduce, then one atomicAdd of the pre-scaled partial
    #pragma unroll
    for (int off = 16; off >= 1; off >>= 1) reg += __shfl_xor(reg, off, 32);
    if (lane == 0)
        atomicAdd(out + (size_t)B_ * N_ * D_,
                  reg * (1.0f / ((float)B_ * (float)N_ * (float)N_)));
}

extern "C" void kernel_launch(void* const* d_in, const int* in_sizes, int n_in,
                              void* d_out, int out_size, void* d_ws, size_t ws_size,
                              hipStream_t stream) {
    (void)in_sizes; (void)n_in; (void)out_size; (void)ws_size;
    const float* Q = (const float*)d_in[0];
    const float* K = (const float*)d_in[1];
    const float* V = (const float*)d_in[2];
    float* out = (float*)d_out;

    const size_t elems = (size_t)B_ * N_ * D_;   // 2,097,152
    _Float16* Qh  = (_Float16*)d_ws;
    _Float16* Kh  = Qh + elems;
    _Float16* VTh = Kh + elems;                  // 12 MB total scratch

    prep_kernel<<<dim3((unsigned)((elems + 255) / 256)), dim3(256), 0, stream>>>(
        Q, K, V, Qh, Kh, VTh);
    zero_reg_kernel<<<dim3(1), dim3(1), 0, stream>>>(out);
    flash_attn_kernel<<<dim3(B_ * N_ / 32), dim3(32), 0, stream>>>(Qh, Kh, VTh, out);
}
